// Attention_63866163692087
// MI455X (gfx1250) — matrix operand order repaired and checked by reading, not compiled, not run on hardware
//
#include <hip/hip_runtime.h>
#include <hip/hip_bf16.h>

// ---------------------------------------------------------------------------
// Problem constants (from reference): B=4, C=512, T=2048, N=4096, H=4, d=128
// All GEMMs are "NT": out[m,n] = dot(Arow_m[K], Brow_n[K]), K = 512.
// Wave tile: 32x64 (2x4 WMMA frags).  Block tile: 128x128 (8 waves, 4x2).
// ---------------------------------------------------------------------------

typedef __attribute__((ext_vector_type(16))) __bf16 v16bf;
typedef __attribute__((ext_vector_type(8)))  __bf16 v8bf;
typedef __attribute__((ext_vector_type(8)))  float  v8f;

// A-fragment (16x32 bf16, MxK). Layout: lane<16 holds row M=lane,
// K = 0..15; lane>=16 holds row M=lane-16, K = 16..31 (matches B).
static __device__ inline v16bf load_frag_a(const __bf16* __restrict__ A,
                                           int row, int K, int k0, int lane) {
  const int half = lane >> 4;
  const __bf16* p = A + (size_t)row * K + k0 + half * 16;
  v8bf lo = *(const v8bf*)p;          // K = half*16 + [0..7]
  v8bf hi = *(const v8bf*)(p + 8);    // K = half*16 + 8 + [0..7]
  return __builtin_shufflevector(lo, hi, 0,1,2,3,4,5,6,7,8,9,10,11,12,13,14,15);
}

// B-fragment (32x16 bf16, KxN). Lanes 0-15: col N=lane, K=0..15;
// lanes 16-31: col N=lane-16, K=16..31. B stored as [N, K] row-major,
// so each lane reads 16 contiguous bf16.
static __device__ inline v16bf load_frag_b(const __bf16* __restrict__ Bm,
                                           int col, int K, int k0, int lane) {
  const int half = lane >> 4;
  const __bf16* p = Bm + (size_t)col * K + k0 + half * 16;
  return *(const v16bf*)p;
}

// MODE 0: f32 store.  MODE 1: bf16 store.
// MODE 2: bf16 store of acc * cg[row][n>>7] * scale  (gated, pre-scaled Q).
template <int MODE>
__global__ __launch_bounds__(256) void gemm_nt_bf16(
    const __bf16* __restrict__ A, const __bf16* __restrict__ Bm,
    void* __restrict__ Cout, const float* __restrict__ cg,
    int K, int ldc, float scale, long long bStrideB, long long bStrideC) {
  const int lane = threadIdx.x & 31;
  const int wave = threadIdx.x >> 5;      // 8 waves
  const int wm = wave >> 1;               // 0..3  -> 32-row slice
  const int wn = wave & 1;                // 0..1  -> 64-col slice
  const int m_base = blockIdx.x * 128 + wm * 32;
  const int n_base = blockIdx.y * 128 + wn * 64;
  const int l15 = lane & 15;
  const int half = lane >> 4;

  const __bf16* Bp = Bm + (size_t)blockIdx.z * (size_t)bStrideB;
  const size_t coff = (size_t)blockIdx.z * (size_t)bStrideC;

  v8f acc[2][4] = {};

  for (int k0 = 0; k0 < K; k0 += 32) {
    v16bf a0 = load_frag_a(A, m_base + l15,      K, k0, lane);
    v16bf a1 = load_frag_a(A, m_base + 16 + l15, K, k0, lane);
    v16bf b0 = load_frag_b(Bp, n_base + l15,      K, k0, lane);
    v16bf b1 = load_frag_b(Bp, n_base + 16 + l15, K, k0, lane);
    v16bf b2 = load_frag_b(Bp, n_base + 32 + l15, K, k0, lane);
    v16bf b3 = load_frag_b(Bp, n_base + 48 + l15, K, k0, lane);
    acc[0][0] = __builtin_amdgcn_wmma_f32_16x16x32_bf16(false, a0, false, b0,
                    (short)0, acc[0][0], false, false);
    acc[0][1] = __builtin_amdgcn_wmma_f32_16x16x32_bf16(false, a0, false, b1,
                    (short)0, acc[0][1], false, false);
    acc[0][2] = __builtin_amdgcn_wmma_f32_16x16x32_bf16(false, a0, false, b2,
                    (short)0, acc[0][2], false, false);
    acc[0][3] = __builtin_amdgcn_wmma_f32_16x16x32_bf16(false, a0, false, b3,
                    (short)0, acc[0][3], false, false);
    acc[1][0] = __builtin_amdgcn_wmma_f32_16x16x32_bf16(false, a1, false, b0,
                    (short)0, acc[1][0], false, false);
    acc[1][1] = __builtin_amdgcn_wmma_f32_16x16x32_bf16(false, a1, false, b1,
                    (short)0, acc[1][1], false, false);
    acc[1][2] = __builtin_amdgcn_wmma_f32_16x16x32_bf16(false, a1, false, b2,
                    (short)0, acc[1][2], false, false);
    acc[1][3] = __builtin_amdgcn_wmma_f32_16x16x32_bf16(false, a1, false, b3,
                    (short)0, acc[1][3], false, false);
  }

  // D layout: n = lane&15, m = r + 8*(lane>>4) within each 16x16 frag.
#pragma unroll
  for (int i = 0; i < 2; ++i) {
#pragma unroll
    for (int j = 0; j < 4; ++j) {
      const int n = n_base + j * 16 + l15;
      const int m_off = m_base + i * 16 + half * 8;
#pragma unroll
      for (int r = 0; r < 8; ++r) {
        const int m_row = m_off + r;
        float v = acc[i][j][r];
        if constexpr (MODE == 0) {
          ((float*)Cout)[coff + (size_t)m_row * ldc + n] = v;
        } else if constexpr (MODE == 1) {
          ((__bf16*)Cout)[coff + (size_t)m_row * ldc + n] = (__bf16)v;
        } else {
          v *= cg[(size_t)m_row * 4 + (n >> 7)] * scale;
          ((__bf16*)Cout)[coff + (size_t)m_row * ldc + n] = (__bf16)v;
        }
      }
    }
  }
}

// Elementwise f32 -> bf16.
__global__ void f32_to_bf16_kernel(const float* __restrict__ x,
                                   __bf16* __restrict__ y, int n) {
  int i = blockIdx.x * blockDim.x + threadIdx.x;
  if (i < n) y[i] = (__bf16)x[i];
}

// Tiled transpose + convert: in[z][r][c] (R x Ccols f32) -> out[z][c*R + r] bf16.
__global__ __launch_bounds__(256) void transpose_f32_bf16(
    const float* __restrict__ in, __bf16* __restrict__ out, int R, int Ccols) {
  __shared__ float tile[32][33];
  const int z = blockIdx.z;
  const int c0 = blockIdx.x * 32;
  const int r0 = blockIdx.y * 32;
  const float* ip = in + (size_t)z * R * Ccols;
  __bf16* op = out + (size_t)z * R * Ccols;
  const int tx = threadIdx.x, ty = threadIdx.y;
  for (int rr = ty; rr < 32; rr += 8)
    tile[rr][tx] = ip[(size_t)(r0 + rr) * Ccols + c0 + tx];
  __syncthreads();
  for (int rr = ty; rr < 32; rr += 8)
    op[(size_t)(c0 + rr) * R + r0 + tx] = (__bf16)tile[tx][rr];
}

// Head-pool gates: cg[bt][h] = sum_c hidden[b][c][t] * Wp[c][h] + bp[h], H=4.
__global__ void cg_kernel(const float* __restrict__ hidden,
                          const float* __restrict__ Wp,
                          const float* __restrict__ bp,
                          float* __restrict__ cg, int Bsz, int C, int T) {
  int bt = blockIdx.x * blockDim.x + threadIdx.x;
  if (bt >= Bsz * T) return;
  int b = bt / T, t = bt - b * T;
  float s0 = bp[0], s1 = bp[1], s2 = bp[2], s3 = bp[3];
  const float* hp = hidden + (size_t)b * C * T + t;
  for (int c = 0; c < C; ++c) {
    float x = hp[(size_t)c * T];
    const float* w = Wp + c * 4;
    s0 += x * w[0]; s1 += x * w[1]; s2 += x * w[2]; s3 += x * w[3];
  }
  float* o = cg + (size_t)bt * 4;
  o[0] = s0; o[1] = s1; o[2] = s2; o[3] = s3;
}

// Per-(b,t) argmax over N (first-max tie break, like jnp.argmax) + value gather.
__global__ __launch_bounds__(256) void argmax_gather(
    const float* __restrict__ logits, const float* __restrict__ Vp,
    float* __restrict__ idx_out, float* __restrict__ zq, int N, int T, int C) {
  const int b = blockIdx.y;
  const int t = blockIdx.x * 256 + threadIdx.x;
  const float* lp = logits + (size_t)b * N * T + t;
  float best = lp[0];
  int bi = 0;
  for (int n = 1; n < N; ++n) {
    float v = lp[(size_t)n * T];
    if (v > best) { best = v; bi = n; }
  }
  idx_out[(size_t)b * T + t] = (float)bi;
  const float* vrow = Vp + (size_t)bi * C;
  float* zp = zq + (size_t)b * C * T + t;
  for (int c = 0; c < C; ++c) zp[(size_t)c * T] = vrow[c];
}

extern "C" void kernel_launch(void* const* d_in, const int* in_sizes, int n_in,
                              void* d_out, int out_size, void* d_ws, size_t ws_size,
                              hipStream_t stream) {
  (void)in_sizes; (void)n_in; (void)out_size; (void)ws_size;
  const int Bb = 4, C = 512, Tt = 2048, Nn = 4096;

  const float* hidden   = (const float*)d_in[0];  // [B,C,T]
  const float* codebook = (const float*)d_in[1];  // [N,C]
  const float* Wq = (const float*)d_in[2];
  const float* Wk = (const float*)d_in[4];
  const float* Wv = (const float*)d_in[6];
  const float* Wp = (const float*)d_in[8];
  const float* bp = (const float*)d_in[9];

  // Workspace layout (~35 MB)
  char* ws = (char*)d_ws;
  __bf16* cb_bf = (__bf16*)ws; ws += (size_t)Nn * C * 2;       // codebook bf16
  __bf16* Wqt   = (__bf16*)ws; ws += (size_t)C * C * 2;        // Wq^T bf16 [out,in]
  __bf16* Wkt   = (__bf16*)ws; ws += (size_t)C * C * 2;
  __bf16* Wvt   = (__bf16*)ws; ws += (size_t)C * C * 2;
  __bf16* hsT   = (__bf16*)ws; ws += (size_t)Bb * Tt * C * 2;  // [B*T, C] bf16
  float*  cgbuf = (float*)ws;  ws += (size_t)Bb * Tt * 4 * 4;  // [B*T, H] f32
  __bf16* qg    = (__bf16*)ws; ws += (size_t)Bb * Tt * C * 2;  // gated scaled Q
  __bf16* Kp    = (__bf16*)ws; ws += (size_t)Nn * C * 2;       // K proj bf16
  float*  Vp    = (float*)ws;                                  // V proj f32

  float* logits = (float*)d_out;                         // [B,N,T]
  float* idxout = logits + (size_t)Bb * Nn * Tt;         // [B,1,T] as float
  float* zq     = idxout + (size_t)Bb * Tt;              // [B,C,T]

  dim3 tb(32, 8);

  // Prep: conversions / transposes
  f32_to_bf16_kernel<<<(Nn * C + 255) / 256, 256, 0, stream>>>(codebook, cb_bf, Nn * C);
  transpose_f32_bf16<<<dim3(C / 32, C / 32, 1), tb, 0, stream>>>(Wq, Wqt, C, C);
  transpose_f32_bf16<<<dim3(C / 32, C / 32, 1), tb, 0, stream>>>(Wk, Wkt, C, C);
  transpose_f32_bf16<<<dim3(C / 32, C / 32, 1), tb, 0, stream>>>(Wv, Wvt, C, C);
  transpose_f32_bf16<<<dim3(Tt / 32, C / 32, Bb), tb, 0, stream>>>(hidden, hsT, C, Tt);

  // Head gates
  cg_kernel<<<(Bb * Tt + 255) / 256, 256, 0, stream>>>(hidden, Wp, bp, cgbuf, Bb, C, Tt);

  // Projections (WMMA): Kp (bf16), Vp (f32).  Block tile 128x128.
  gemm_nt_bf16<1><<<dim3(Nn / 128, C / 128, 1), 256, 0, stream>>>(
      cb_bf, Wkt, Kp, nullptr, C, C, 1.0f, 0, 0);
  gemm_nt_bf16<0><<<dim3(Nn / 128, C / 128, 1), 256, 0, stream>>>(
      cb_bf, Wvt, Vp, nullptr, C, C, 1.0f, 0, 0);

  // Gated, pre-scaled Q: qg = (hsT @ Wq^T) * cg[h] / (sqrt(128)*sqrt(4))
  const float kscale = 0.04419417382415922f;
  gemm_nt_bf16<2><<<dim3(Bb * Tt / 128, C / 128, 1), 256, 0, stream>>>(
      hsT, Wqt, qg, cgbuf, C, C, kscale, 0, 0);

  // logits[b] = Kp [4096,512] x qg_b [2048,512]^T  -> [N, T] f32 into d_out
  gemm_nt_bf16<0><<<dim3(Nn / 128, Tt / 128, Bb), 256, 0, stream>>>(
      Kp, qg, logits, nullptr, C, Tt, 1.0f,
      (long long)Tt * C, (long long)Nn * Tt);

  // argmax over N + z_q gather (logits are L2-resident: 134 MB < 192 MB L2)
  argmax_gather<<<dim3(Tt / 256, Bb), 256, 0, stream>>>(logits, Vp, idxout, zq, Nn, Tt, C);
}
